// CrossAttention_44693429682227
// MI455X (gfx1250) — matrix operand order repaired and checked by reading, not compiled, not run on hardware
//
#include <hip/hip_runtime.h>

#define HEADS 8
#define EMB 128
#define NC 32              // N-chunks over the key bank

typedef __attribute__((ext_vector_type(16))) _Float16 v16h;
typedef __attribute__((ext_vector_type(8)))  float    v8f;

union Frag16 {
    v16h v;
    uint4 u[2];
};

// LDS layout (single shared block, byte offsets). Stage = 32 keys.
#define KROW_BYTES 288      // 144 f16 per row: bank-spread padding, 16B aligned
#define KBUF_BYTES (32 * KROW_BYTES)   // 9216
#define KBUF0 0
#define KBUF1 KBUF_BYTES
#define VBUF0 (2 * KBUF_BYTES)         // 18432
#define VBUF1 (VBUF0 + 128)
#define SMEM_BYTES (VBUF1 + 128)       // 18688

// ---------------------------------------------------------------------------
// CDNA5 async global->LDS copies (ASYNCcnt) via inline asm.
// LDS byte address = low 32 bits of the generic shared-aperture pointer.
// ---------------------------------------------------------------------------
__device__ __forceinline__ void async_b128(const void* g, void* l)
{
    asm volatile("global_load_async_to_lds_b128 %0, %1, off"
                 :: "v"((unsigned)(uintptr_t)l),
                    "v"((unsigned long long)(uintptr_t)g)
                 : "memory");
}
__device__ __forceinline__ void async_b32(const void* g, void* l)
{
    asm volatile("global_load_async_to_lds_b32 %0, %1, off"
                 :: "v"((unsigned)(uintptr_t)l),
                    "v"((unsigned long long)(uintptr_t)g)
                 : "memory");
}
__device__ __forceinline__ void wait_async0()
{
    asm volatile("s_wait_asynccnt 0x0" ::: "memory");
}

// ---------------------------------------------------------------------------
// Key bank f32 -> f16 pre-conversion (one pass; later re-read from L2 at half
// the bytes; removes all cvt work from the attention hot loop).
// ---------------------------------------------------------------------------
__global__ void kconv_kernel(const float* __restrict__ src,
                             _Float16* __restrict__ dst, long n)
{
    const long i = ((long)blockIdx.x * blockDim.x + threadIdx.x) * 8;
    if (i + 7 < n) {
        const float4 a = *(const float4*)(src + i);
        const float4 b = *(const float4*)(src + i + 4);
        union { _Float16 h[8]; uint4 u; } o;
        o.h[0] = (_Float16)a.x; o.h[1] = (_Float16)a.y;
        o.h[2] = (_Float16)a.z; o.h[3] = (_Float16)a.w;
        o.h[4] = (_Float16)b.x; o.h[5] = (_Float16)b.y;
        o.h[6] = (_Float16)b.z; o.h[7] = (_Float16)b.w;
        *(uint4*)(dst + i) = o.u;
    } else {
        for (long j = i; j < n; ++j) dst[j] = (_Float16)src[j];
    }
}

// ---------------------------------------------------------------------------
// Phase 1: q = (x @ Wq^T) * EMB^-0.25, f32 inputs -> f16 output in workspace.
// One wave per 16x16 output tile, chained v_wmma_f32_16x16x32_f16 over K.
// ---------------------------------------------------------------------------
__global__ void qproj_kernel(const float* __restrict__ x,
                             const float* __restrict__ Wq,
                             _Float16* __restrict__ qf16,
                             int M, int K, int Ncols)
{
    const int lane = threadIdx.x & 31;
    const int wave = threadIdx.x >> 5;
    const int half = lane >> 4;
    const int l16  = lane & 15;

    const int ntiles = Ncols >> 4;
    const int tile  = blockIdx.x * 8 + wave;
    const int mtile = tile / ntiles;
    const int ntile = tile - mtile * ntiles;
    if (mtile * 16 >= M) return;

    const float* xrow = x  + (size_t)(mtile * 16 + l16) * K;
    const float* wrow = Wq + (size_t)(ntile * 16 + l16) * K;  // B[k][o] = Wq[o][k]

    v8f acc = {};
    for (int k0 = 0; k0 < K; k0 += 32) {
        Frag16 a, b;
        #pragma unroll
        for (int j = 0; j < 8; ++j) {
            a.v[j]     = (_Float16)xrow[k0 + half * 8 + j];
            a.v[j + 8] = (_Float16)xrow[k0 + 16 + half * 8 + j];
            b.v[j]     = (_Float16)wrow[k0 + half * 8 + j];
            b.v[j + 8] = (_Float16)wrow[k0 + 16 + half * 8 + j];
        }
        acc = __builtin_amdgcn_wmma_f32_16x16x32_f16(false, a.v, false, b.v,
                                                     (short)0, acc, false, false);
    }

    const float scale = 0.29730177875068026f;  // 128^-0.25
    #pragma unroll
    for (int r = 0; r < 8; ++r) {
        const int m = mtile * 16 + r + half * 8;
        qf16[(size_t)m * Ncols + ntile * 16 + l16] = (_Float16)(acc[r] * scale);
    }
}

// ---------------------------------------------------------------------------
// Stage one 32-key stage (f16 rows) + 32 values into an LDS buffer with async
// DMA: 256 threads x two b128 each (8KB), values via b32.
// ---------------------------------------------------------------------------
__device__ __forceinline__ void stage_keys(const _Float16* __restrict__ kh,
                                           const float* __restrict__ vals,
                                           unsigned char* smem,
                                           int tid, int base, int Nkeys, int bufsel)
{
    const int row = tid >> 4;   // 0..15
    const int blk = tid & 15;   // 16B block within the 256B key row
    unsigned char* kb = smem + (bufsel ? KBUF1 : KBUF0);
    #pragma unroll
    for (int h32 = 0; h32 < 2; ++h32) {
        const int r = row + h32 * 16;
        if (base + r < Nkeys)
            async_b128(kh + (size_t)(base + r) * 128 + blk * 8,
                       kb + r * KROW_BYTES + blk * 16);
    }
    if (tid < 32 && base + tid < Nkeys)
        async_b32(vals + base + tid,
                  smem + (bufsel ? VBUF1 : VBUF0) + tid * 4);
}

// ---------------------------------------------------------------------------
// Phase 2: streaming softmax-attention. 8 waves/block, each wave owns 32 rows
// (2 M-tiles); block shares double-buffered async-staged 32-key stages.
// Per stage: 16 WMMAs per wave (2 sub-tiles x 4 K-chunks x 2 M-tiles) behind
// a single s_wait_asynccnt + barrier; exp/accumulate without max subtraction
// (scores bounded << 88).
// ---------------------------------------------------------------------------
__global__ void attn_kernel(const _Float16* __restrict__ qf16,
                            const _Float16* __restrict__ kh,
                            const float* __restrict__ values,
                            float* __restrict__ partA,
                            float* __restrict__ partL,
                            int Nkeys, int totRows)
{
    __shared__ __align__(16) unsigned char smem[SMEM_BYTES];

    const int tid  = threadIdx.x;
    const int lane = tid & 31;
    const int wave = tid >> 5;
    const int half = lane >> 4;
    const int ncol = lane & 15;

    const int chunk = blockIdx.x;
    const int W0    = (blockIdx.y * 8 + wave) * 32;   // first of this wave's 32 rows

    // Preload Q fragments: 2 M-tiles x 4 K-chunks (layout per ISA A-matrix 16x32).
    Frag16 aq[2][4];
    #pragma unroll
    for (int mt = 0; mt < 2; ++mt) {
        const _Float16* qrow = qf16 + (size_t)(W0 + mt * 16 + ncol) * 128;
        #pragma unroll
        for (int c = 0; c < 4; ++c) {
            const uint4* p = (const uint4*)(qrow + c * 32 + half * 8);
            aq[mt][c].u[0] = p[0];   // j=0..7  : K = c*32 + half*8 + j
            aq[mt][c].u[1] = p[2];   // j=8..15 : K = c*32 + 16 + half*8 + j
        }
    }

    v8f Lrun[2] = {{}, {}};
    v8f Arun[2] = {{}, {}};

    const int totStages = (Nkeys + 31) >> 5;
    const int spc = (totStages + (int)gridDim.x - 1) / (int)gridDim.x;
    const int s0 = chunk * spc;
    int s1 = s0 + spc;
    if (s1 > totStages) s1 = totStages;

    if (s0 < s1) stage_keys(kh, values, smem, tid, s0 << 5, Nkeys, s0 & 1);

    for (int s = s0; s < s1; ++s) {
        const int sel = s & 1;

        wait_async0();       // this wave's outstanding copies (incl. buf[sel]) done
        __syncthreads();     // ...and everyone else's; prev compute also done

        if (s + 1 < s1)      // overlap next stage's DMA with this stage's WMMAs
            stage_keys(kh, values, smem, tid, (s + 1) << 5, Nkeys, (s + 1) & 1);

        const unsigned char* kb = smem + (sel ? KBUF1 : KBUF0);
        const float*         vb = (const float*)(smem + (sel ? VBUF1 : VBUF0));

        #pragma unroll
        for (int sub = 0; sub < 2; ++sub) {
            const int base = (s << 5) + sub * 16;
            const unsigned char* krow = kb + (sub * 16 + ncol) * KROW_BYTES;

            // Scores: 32 rows x 16 keys, K=128. B fragment reused by both M-tiles.
            v8f sc0 = {}, sc1 = {};
            #pragma unroll
            for (int c = 0; c < 4; ++c) {
                Frag16 bk;
                const uint4* p = (const uint4*)(krow + c * 64 + half * 16);
                bk.u[0] = p[0];
                bk.u[1] = p[2];
                sc0 = __builtin_amdgcn_wmma_f32_16x16x32_f16(false, aq[0][c].v, false,
                                                             bk.v, (short)0, sc0,
                                                             false, false);
                sc1 = __builtin_amdgcn_wmma_f32_16x16x32_f16(false, aq[1][c].v, false,
                                                             bk.v, (short)0, sc1,
                                                             false, false);
            }

            const bool  ok = (base + ncol) < Nkeys;
            const float vv = vb[sub * 16 + ncol];
            #pragma unroll
            for (int r = 0; r < 8; ++r) {
                const float w0 = ok ? __expf(sc0[r]) : 0.f;
                const float w1 = ok ? __expf(sc1[r]) : 0.f;
                Lrun[0][r] += w0;  Arun[0][r] += w0 * vv;
                Lrun[1][r] += w1;  Arun[1][r] += w1 * vv;
            }
        }
    }

    // Column reduction: each row lives across the 16 lanes of one half;
    // xor masks 1..8 stay within the half.
    #pragma unroll
    for (int mt = 0; mt < 2; ++mt) {
        #pragma unroll
        for (int r = 0; r < 8; ++r) {
            float Lr = Lrun[mt][r];
            float Ar = Arun[mt][r];
            for (int sh = 8; sh > 0; sh >>= 1) {
                Lr += __shfl_xor(Lr, sh, 32);
                Ar += __shfl_xor(Ar, sh, 32);
            }
            if (ncol == 0) {
                const int row = W0 + mt * 16 + r + half * 8;
                partL[(size_t)chunk * totRows + row] = Lr;
                partA[(size_t)chunk * totRows + row] = Ar;
            }
        }
    }
}

// ---------------------------------------------------------------------------
// Phase 3: combine partials, divide, mean over heads, add curiosity.
// ---------------------------------------------------------------------------
__global__ void combine_kernel(const float* __restrict__ partA,
                               const float* __restrict__ partL,
                               const float* __restrict__ cur,
                               float* __restrict__ out,
                               int BT, int nchunks, int totRows)
{
    const int bt = blockIdx.x * blockDim.x + threadIdx.x;
    if (bt >= BT) return;
    float s = 0.f;
    for (int h = 0; h < HEADS; ++h) {
        const int row = bt * HEADS + h;
        float A = 0.f, L = 0.f;
        for (int c = 0; c < nchunks; ++c) {
            A += partA[(size_t)c * totRows + row];
            L += partL[(size_t)c * totRows + row];
        }
        s += A / L;
    }
    out[bt] = s * (1.0f / HEADS) + cur[bt];
}

extern "C" void kernel_launch(void* const* d_in, const int* in_sizes, int n_in,
                              void* d_out, int out_size, void* d_ws, size_t ws_size,
                              hipStream_t stream)
{
    const float* x    = (const float*)d_in[0];
    const float* cur  = (const float*)d_in[1];
    const float* Wq   = (const float*)d_in[2];
    const float* keys = (const float*)d_in[3];
    const float* vals = (const float*)d_in[4];

    const int Ncols   = HEADS * EMB;            // 1024
    const int hin     = in_sizes[2] / Ncols;    // 1024
    const int BT      = in_sizes[0] / hin;      // 512
    const int Nkeys   = in_sizes[4];            // 50000
    const int totRows = BT * HEADS;             // 4096

    // Workspace: qf16 | keys-f16 | partA | partL
    char* ws = (char*)d_ws;
    _Float16* qf16 = (_Float16*)ws;
    size_t off = ((size_t)BT * Ncols * sizeof(_Float16) + 255) & ~(size_t)255;
    _Float16* kh = (_Float16*)(ws + off);
    off += ((size_t)Nkeys * EMB * sizeof(_Float16) + 255) & ~(size_t)255;
    float* partA = (float*)(ws + off);
    float* partL = partA + (size_t)NC * totRows;

    // Key bank pre-conversion f32 -> f16
    {
        const long n = (long)Nkeys * EMB;
        const long threads = (n + 7) / 8;
        kconv_kernel<<<dim3((unsigned)((threads + 255) / 256)), dim3(256), 0, stream>>>(
            keys, kh, n);
    }

    // Phase 1: Q projection
    const int waveTiles = (BT / 16) * (Ncols / 16);
    qproj_kernel<<<dim3((waveTiles + 7) / 8), dim3(256), 0, stream>>>(
        x, Wq, qf16, BT, hin, Ncols);

    // Phase 2: attention sweep (NC chunks x 16 M-groups, 8 waves x 32 rows each)
    attn_kernel<<<dim3(NC, totRows / 256), dim3(256), 0, stream>>>(
        qf16, kh, vals, partA, partL, Nkeys, totRows);

    // Phase 3: combine
    combine_kernel<<<dim3((BT + 255) / 256), dim3(256), 0, stream>>>(
        partA, partL, cur, (float*)d_out, BT, NC, totRows);
}
